// FixedGeometryGNN_7069516169832
// MI455X (gfx1250) — matrix operand order repaired and checked by reading, not compiled, not run on hardware
//
#include <hip/hip_runtime.h>

// ---------------------------------------------------------------------------
// GCN (3 layers, D=128) for gfx1250.
// GEMM via v_wmma_f32_16x16x32_f16 (f16 in, f32 accumulate).
// Aggregation via L2-resident gather + global_atomic_add_f32 scatter.
// ---------------------------------------------------------------------------

typedef __attribute__((ext_vector_type(16))) _Float16 v16h;
typedef __attribute__((ext_vector_type(8)))  _Float16 v8h;
typedef __attribute__((ext_vector_type(8)))  float    v8f;

#define GCN_N 100000
#define GCN_E 1600000
#define GCN_D 128
#define LDS_STRIDE 136   // 16B-aligned rows, 4-bank step per row -> conflict-lite

// ---- degree / norm precompute ---------------------------------------------

__global__ void deg_init_kernel(unsigned* __restrict__ deg, int n) {
  int i = blockIdx.x * blockDim.x + threadIdx.x;
  if (i < n) deg[i] = 1u;  // self-loop
}

__global__ void deg_accum_kernel(const int* __restrict__ ei, unsigned* __restrict__ deg, int ne) {
  int e = blockIdx.x * blockDim.x + threadIdx.x;
  if (e < ne) atomicAdd(&deg[ei[ne + e]], 1u);  // dst row is second half
}

__global__ void dinv_kernel(unsigned* __restrict__ degdinv, int n) {
  int i = blockIdx.x * blockDim.x + threadIdx.x;
  if (i < n) {
    float d = (float)degdinv[i];
    degdinv[i] = __float_as_uint(rsqrtf(d));  // in-place uint -> float
  }
}

__global__ void norm_kernel(const int* __restrict__ ei, const float* __restrict__ dinv,
                            float* __restrict__ norm, int ne) {
  int e = blockIdx.x * blockDim.x + threadIdx.x;
  if (e < ne) norm[e] = dinv[ei[e]] * dinv[ei[ne + e]];
}

// ---- weight permutation into WMMA B-fragment order ------------------------
// wperm[(((nt*4)+kc)*32 + lane)*16 + e] = f16( W[K][Ncol] )
//   K = kc*32 + (lane<16?0:8) + (e<8 ? e : 16+e-8) ; Ncol = nt*16 + (lane&15)
__global__ void permute_w_kernel(const float* __restrict__ W, _Float16* __restrict__ wperm) {
  int t = blockIdx.x * blockDim.x + threadIdx.x;
  if (t >= GCN_D * GCN_D) return;
  int e    = t & 15;
  int lane = (t >> 4) & 31;
  int kc   = (t >> 9) & 3;
  int nt   = t >> 11;
  int K    = kc * 32 + ((lane < 16) ? 0 : 8) + ((e < 8) ? e : 16 + (e - 8));
  int Ncol = nt * 16 + (lane & 15);
  wperm[t] = (_Float16)W[K * GCN_D + Ncol];
}

// ---- dense transform: h = (relu?)x @ W  -----------------------------------
// One block = 16 rows x 128 cols. 8 waves; wave w owns N-tile w. K-loop of 4.
__global__ __launch_bounds__(256) void gemm_wmma_kernel(
    const float* __restrict__ xin, const _Float16* __restrict__ wperm,
    float* __restrict__ hout, int applyRelu) {
  __shared__ _Float16 ldsA[16 * LDS_STRIDE];
  const int t = threadIdx.x;
  const int rowBase = blockIdx.x * 16;  // N=100000 is divisible by 16

  // stage A tile (16x128 f32 -> f16, optional ReLU), coalesced
#pragma unroll
  for (int i = 0; i < 8; ++i) {
    int g = t + i * 256;           // 0..2047
    int r = g >> 7, c2 = g & 127;
    float v = xin[(size_t)(rowBase + r) * GCN_D + c2];
    if (applyRelu) v = fmaxf(v, 0.0f);
    ldsA[r * LDS_STRIDE + c2] = (_Float16)v;
  }
  __syncthreads();

  const int lane = t & 31;
  const int wv   = t >> 5;               // 0..7: N-tile index
  const int m    = lane & 15;            // A row within tile
  const int klo  = (lane < 16) ? 0 : 8;  // A K sub-offset per ISA layout
  const int nBase = wv * 16;

  v8f c = {};
#pragma unroll
  for (int kc = 0; kc < 4; ++kc) {
    // A fragment: elements 0..7 -> K = kc*32+klo+0..7 ; 8..15 -> +16..+23
    const _Float16* pa = ldsA + m * LDS_STRIDE + kc * 32 + klo;
    v8h lo = *(const v8h*)pa;
    v8h hi = *(const v8h*)(pa + 16);
    v16h a = __builtin_shufflevector(lo, hi, 0, 1, 2, 3, 4, 5, 6, 7,
                                     8, 9, 10, 11, 12, 13, 14, 15);
    // B fragment: one contiguous 32B load per lane (pre-permuted)
    v16h b = *(const v16h*)(wperm + (((size_t)wv * 4 + kc) * 32 + lane) * 16);
    c = __builtin_amdgcn_wmma_f32_16x16x32_f16(false, a, false, b,
                                               (short)0, c, false, false);
  }

  // C/D layout: VGPR r -> M = r + (lane<16?0:8), N = lane&15
  const int rOff = (lane >> 4) << 3;
  float* outp = hout + (size_t)(rowBase + rOff) * GCN_D + nBase + (lane & 15);
#pragma unroll
  for (int r = 0; r < 8; ++r) outp[(size_t)r * GCN_D] = c[r];
}

// ---- out = bias + dinv^2 * h  (self-loop message + bias init) -------------
__global__ void init_out_kernel(const float* __restrict__ h, const float* __restrict__ dinv,
                                const float* __restrict__ bias, float* __restrict__ out, int n) {
  int t = blockIdx.x * blockDim.x + threadIdx.x;
  int i = t >> 5, q = t & 31;
  if (i >= n) return;
  float di = dinv[i];
  float s = di * di;
  float4 hv = ((const float4*)(h + (size_t)i * GCN_D))[q];
  float4 bv = ((const float4*)bias)[q];
  float4 o = make_float4(bv.x + s * hv.x, bv.y + s * hv.y,
                         bv.z + s * hv.z, bv.w + s * hv.w);
  ((float4*)(out + (size_t)i * GCN_D))[q] = o;
}

// ---- edge scatter: out[dst] += norm[e] * h[src]  --------------------------
// One wave per edge: coalesced 512B gather of h[src], 4 f32 atomics per lane.
// h (51.2MB) is L2-resident on MI455X (192MB L2), so this stays on-die.
__global__ __launch_bounds__(256) void scatter_kernel(
    const float* __restrict__ h, const int* __restrict__ ei,
    const float* __restrict__ norm, float* __restrict__ out, int ne) {
  int t = blockIdx.x * 256 + threadIdx.x;
  int e = t >> 5, q = t & 31;
  if (e >= ne) return;
  int s = ei[e];
  int d = ei[ne + e];
  float w = norm[e];
  float4 hv = ((const float4*)(h + (size_t)s * GCN_D))[q];
  float* o = out + (size_t)d * GCN_D + q * 4;
  atomicAdd(o + 0, hv.x * w);
  atomicAdd(o + 1, hv.y * w);
  atomicAdd(o + 2, hv.z * w);
  atomicAdd(o + 3, hv.w * w);
}

// ---------------------------------------------------------------------------

extern "C" void kernel_launch(void* const* d_in, const int* in_sizes, int n_in,
                              void* d_out, int out_size, void* d_ws, size_t ws_size,
                              hipStream_t stream) {
  const float* x  = (const float*)d_in[0];
  const int*   ei = (const int*)d_in[1];   // [2][E] int32: row0=src, row1=dst
  const float* W1 = (const float*)d_in[2];
  const float* b1 = (const float*)d_in[3];
  const float* W2 = (const float*)d_in[4];
  const float* b2 = (const float*)d_in[5];
  const float* W3 = (const float*)d_in[6];
  const float* b3 = (const float*)d_in[7];
  float* out = (float*)d_out;

  const int N = GCN_N, E = GCN_E;

  // carve workspace (256B aligned)
  char* ws = (char*)d_ws;
  size_t off = 0;
  auto carve = [&](size_t bytes) -> void* {
    void* p = ws + off;
    off += (bytes + 255) & ~(size_t)255;
    return p;
  };
  unsigned*  degdinv = (unsigned*)carve((size_t)N * 4);          // deg -> dinv in place
  float*     norm    = (float*)carve((size_t)E * 4);
  _Float16*  wp1     = (_Float16*)carve((size_t)GCN_D * GCN_D * 2);
  _Float16*  wp2     = (_Float16*)carve((size_t)GCN_D * GCN_D * 2);
  _Float16*  wp3     = (_Float16*)carve((size_t)GCN_D * GCN_D * 2);
  float*     bufA    = (float*)carve((size_t)N * GCN_D * 4);     // gemm output
  float*     bufB    = (float*)carve((size_t)N * GCN_D * 4);     // aggregated output
  float*     dinv    = (float*)degdinv;

  const int gN   = (N + 255) / 256;
  const int gE   = (E + 255) / 256;
  const int gWp  = (GCN_D * GCN_D + 255) / 256;
  const int gGem = N / 16;                 // 6250
  const int gIni = (N * 32 + 255) / 256;   // float4 per thread
  const int gSct = (E * 32 + 255) / 256;   // wave per edge

  // normalization precompute
  deg_init_kernel<<<gN, 256, 0, stream>>>(degdinv, N);
  deg_accum_kernel<<<gE, 256, 0, stream>>>(ei, degdinv, E);
  dinv_kernel<<<gN, 256, 0, stream>>>(degdinv, N);
  norm_kernel<<<gE, 256, 0, stream>>>(ei, dinv, norm, E);

  // weights -> f16 fragment order
  permute_w_kernel<<<gWp, 256, 0, stream>>>(W1, wp1);
  permute_w_kernel<<<gWp, 256, 0, stream>>>(W2, wp2);
  permute_w_kernel<<<gWp, 256, 0, stream>>>(W3, wp3);

  // layer 1: x -> bufA -> bufB   (relu deferred to next gemm load)
  gemm_wmma_kernel<<<gGem, 256, 0, stream>>>(x, wp1, bufA, 0);
  init_out_kernel<<<gIni, 256, 0, stream>>>(bufA, dinv, b1, bufB, N);
  scatter_kernel<<<gSct, 256, 0, stream>>>(bufA, ei, norm, bufB, E);

  // layer 2: relu(bufB) -> bufA -> bufB
  gemm_wmma_kernel<<<gGem, 256, 0, stream>>>(bufB, wp2, bufA, 1);
  init_out_kernel<<<gIni, 256, 0, stream>>>(bufA, dinv, b2, bufB, N);
  scatter_kernel<<<gSct, 256, 0, stream>>>(bufA, ei, norm, bufB, E);

  // layer 3: relu(bufB) -> bufA -> d_out  (no final relu)
  gemm_wmma_kernel<<<gGem, 256, 0, stream>>>(bufB, wp3, bufA, 1);
  init_out_kernel<<<gIni, 256, 0, stream>>>(bufA, dinv, b3, out, N);
  scatter_kernel<<<gSct, 256, 0, stream>>>(bufA, ei, norm, out, E);
}
